// RegionProposalNetwork_1546188226982
// MI455X (gfx1250) — compile-verified, matrix operand order checked by reading, not run on hardware
//
#include <hip/hip_runtime.h>
#include <hip/hip_bf16.h>

typedef __attribute__((ext_vector_type(16))) _Float16 v16h;
typedef __attribute__((ext_vector_type(8)))  _Float16 v8h;
typedef __attribute__((ext_vector_type(8)))  float    v8f;

#define N_ANCH 9207
#define NPOS 128
#define NNEG 128
#define TOPN 2000

// ---------------- helpers ----------------

__device__ __forceinline__ int hpos_idx(int l, int b, int p) {
  const int PB[5]  = {0, 4608, 5760, 6048, 6120};
  const int SPL[5] = {2304, 576, 144, 36, 9};
  return PB[l] + b * SPL[l] + p;
}

struct AnchorInfo { int l, p, a; float x1, y1, x2, y2; };

__device__ __forceinline__ AnchorInfo anchor_of(int g) {
  const int   AOFF[6] = {0, 6912, 8640, 9072, 9180, 9207};
  const int   Wl[5]   = {48, 24, 12, 6, 3};
  const float STRD[5] = {4.f, 8.f, 16.f, 32.f, 64.f};
  AnchorInfo ai;
  int l = 0;
#pragma unroll
  for (int i = 1; i < 5; ++i) if (g >= AOFF[i]) l = i;
  ai.l = l;
  int r = g - AOFF[l];
  ai.p = r / 3; ai.a = r - ai.p * 3;
  int W = Wl[l];
  int y = ai.p / W, x = ai.p - y * W;
  float s = STRD[l];
  float side = 32.f * (float)(1 << l);
  float area = side * side;
  float ratio = (ai.a == 0) ? 0.5f : ((ai.a == 1) ? 1.f : 2.f);
  float ws = sqrtf(area / ratio), hs = ws * ratio;
  float cx = ((float)x + 0.5f) * s, cy = ((float)y + 0.5f) * s;
  ai.x1 = cx - ws * 0.5f; ai.y1 = cy - hs * 0.5f;
  ai.x2 = cx + ws * 0.5f; ai.y2 = cy + hs * 0.5f;
  return ai;
}

__device__ __forceinline__ float iou_xyxy(float ax1, float ay1, float ax2, float ay2,
                                          float bx1, float by1, float bx2, float by2) {
  float aa = (ax2 - ax1) * (ay2 - ay1);
  float ab = (bx2 - bx1) * (by2 - by1);
  float lx = fmaxf(ax1, bx1), ly = fmaxf(ay1, by1);
  float rx = fminf(ax2, bx2), ry = fminf(ay2, by2);
  float w = fmaxf(rx - lx, 0.f), h = fmaxf(ry - ly, 0.f);
  float inter = w * h;
  return inter / (aa + ab - inter);
}

// Load the documented 16-bit WMMA per-lane K pattern: elements [0..7] = K kb+hioff..+7,
// elements [8..15] = K kb+16+hioff..+7.  Both chunks are 16B-aligned -> global_load_b128.
__device__ __forceinline__ v16h load_dualrow(const _Float16* __restrict__ base, int kb, int hioff) {
  v8h x = *(const v8h*)(base + kb + hioff);
  v8h y = *(const v8h*)(base + kb + 16 + hioff);
  v16h r;
#pragma unroll
  for (int e = 0; e < 8; ++e) { r[e] = x[e]; r[e + 8] = y[e]; }
  return r;
}

// ---------------- init / weight conversion ----------------

__global__ void init_k(unsigned* gmax, float* accum) {
  int t = threadIdx.x;
  if (t < 40) gmax[t] = 0u;
  if (t < 4)  accum[t] = 0.f;
}

__global__ void cvtw_k(const float* __restrict__ conv_w, const float* __restrict__ cls_w,
                       const float* __restrict__ reg_w, _Float16* __restrict__ wA,
                       _Float16* __restrict__ wCR) {
  int id = blockIdx.x * blockDim.x + threadIdx.x;
  if (id < 256 * 2304) wA[id] = (_Float16)conv_w[id];
  if (id < 16 * 256) {
    int m = id >> 8, k = id & 255;
    float v = 0.f;
    if (m < 3) v = cls_w[m * 256 + k];
    else if (m < 15) v = reg_w[(m - 3) * 256 + k];
    wCR[id] = (_Float16)v;
  }
}

// ---------------- im2col: feat (NCHW f32) -> xcol[n][2304] f16 ----------------
// All the k -> (ci,ky,kx) math and boundary tests live here, once, outside the GEMM loop.

__global__ void im2col_k(const float* __restrict__ feat, _Float16* __restrict__ xcol,
                         int H, int W, int posBase) {
  const int SPL = H * W;
  const long long total = (long long)2 * SPL * 2304;
  for (long long id = (long long)blockIdx.x * blockDim.x + threadIdx.x; id < total;
       id += (long long)gridDim.x * blockDim.x) {
    int n = (int)(id / 2304), k = (int)(id - (long long)n * 2304);
    int b = n / SPL, p = n - b * SPL;
    int py = p / W, px = p - py * W;
    int ci = k / 9, r = k - ci * 9;
    int ky = r / 3, kx = r - ky * 3;
    int iy = py + ky - 1, ix = px + kx - 1;
    float v = 0.f;
    if (iy >= 0 && iy < H && ix >= 0 && ix < W)
      v = feat[(size_t)b * 256 * SPL + (size_t)ci * SPL + iy * W + ix];
    xcol[(size_t)(posBase + n) * 2304 + k] = (_Float16)v;
  }
}

// ---------------- conv3x3 + ReLU: pure WMMA GEMM, 4 M-tiles per wave ----------------
// A = wA (256 x 2304 f16), B = xcol (2304 x N), D = relu(A.B + bias) -> xbuf f16 [pos][256]

__global__ void conv3_wmma(const _Float16* __restrict__ xcol, const _Float16* __restrict__ wA,
                           const float* __restrict__ convb, _Float16* __restrict__ xbuf,
                           int SPL, int posBase) {
  const int Nlvl = 2 * SPL;
  const int nTilesN = (Nlvl + 15) >> 4;
  int wave = (blockIdx.x * blockDim.x + threadIdx.x) >> 5;
  int lane = threadIdx.x & 31;
  int mg = wave & 3, nt = wave >> 2;      // 4 M-groups of 64 rows each
  if (nt >= nTilesN) return;
  int n0 = nt * 16;
  int laneM = lane & 15, hi = lane >> 4;
  int hioff = hi * 8;

  // B column for this lane; out-of-range columns clamp to a valid row: their garbage
  // only affects D columns we never store (GEMM columns are independent).
  int nIdx = n0 + laneM;
  bool nvalid = nIdx < Nlvl;
  int nc = nvalid ? nIdx : (Nlvl - 1);
  const _Float16* brow = xcol + (size_t)(posBase + nc) * 2304;

  const _Float16* a0 = wA + (size_t)(mg * 64 + 0 * 16 + laneM) * 2304;
  const _Float16* a1 = wA + (size_t)(mg * 64 + 1 * 16 + laneM) * 2304;
  const _Float16* a2 = wA + (size_t)(mg * 64 + 2 * 16 + laneM) * 2304;
  const _Float16* a3 = wA + (size_t)(mg * 64 + 3 * 16 + laneM) * 2304;

  v8f acc0 = {}, acc1 = {}, acc2 = {}, acc3 = {};
  for (int kb = 0; kb < 2304; kb += 32) {
    __builtin_prefetch(brow + kb + 256, 0, 0);       // global_prefetch_b8: cover next K slices
    v16h bv = load_dualrow(brow, kb, hioff);
    v16h av0 = load_dualrow(a0, kb, hioff);
    v16h av1 = load_dualrow(a1, kb, hioff);
    v16h av2 = load_dualrow(a2, kb, hioff);
    v16h av3 = load_dualrow(a3, kb, hioff);
    acc0 = __builtin_amdgcn_wmma_f32_16x16x32_f16(false, av0, false, bv, (short)0, acc0, false, false);
    acc1 = __builtin_amdgcn_wmma_f32_16x16x32_f16(false, av1, false, bv, (short)0, acc1, false, false);
    acc2 = __builtin_amdgcn_wmma_f32_16x16x32_f16(false, av2, false, bv, (short)0, acc2, false, false);
    acc3 = __builtin_amdgcn_wmma_f32_16x16x32_f16(false, av3, false, bv, (short)0, acc3, false, false);
  }
  if (nvalid) {
    int bimg = nIdx / SPL, p = nIdx - bimg * SPL;
    _Float16* xr = xbuf + (size_t)(posBase + bimg * SPL + p) * 256;
#pragma unroll
    for (int v = 0; v < 8; ++v) {
      int Mb = mg * 64 + hi * 8 + v;
      float v0 = acc0[v] + convb[Mb + 0];
      float v1 = acc1[v] + convb[Mb + 16];
      float v2 = acc2[v] + convb[Mb + 32];
      float v3 = acc3[v] + convb[Mb + 48];
      xr[Mb + 0]  = (_Float16)(v0 > 0.f ? v0 : 0.f);
      xr[Mb + 16] = (_Float16)(v1 > 0.f ? v1 : 0.f);
      xr[Mb + 32] = (_Float16)(v2 > 0.f ? v2 : 0.f);
      xr[Mb + 48] = (_Float16)(v3 > 0.f ? v3 : 0.f);
    }
  }
}

// ---------------- fused cls(3)+reg(12) 1x1 heads: one 16-row WMMA GEMM ----------------

__global__ void head_wmma(const _Float16* __restrict__ xbuf, const _Float16* __restrict__ wCR,
                          const float* __restrict__ clsb, const float* __restrict__ regb,
                          float* __restrict__ headout, int SPL, int posBase) {
  const int Nlvl = 2 * SPL;
  const int nTiles = (Nlvl + 15) >> 4;
  int wave = (blockIdx.x * blockDim.x + threadIdx.x) >> 5;
  int lane = threadIdx.x & 31;
  if (wave >= nTiles) return;
  int n0 = wave * 16;
  int laneM = lane & 15, hi = lane >> 4;
  int hioff = hi * 8;

  const _Float16* arow = wCR + (size_t)laneM * 256;
  int nIdx = n0 + laneM;
  bool nvalid = nIdx < Nlvl;
  int nc = nvalid ? nIdx : (Nlvl - 1);
  const _Float16* xr = xbuf + (size_t)(posBase + nc) * 256;

  v8f acc = {};
#pragma unroll
  for (int kb = 0; kb < 256; kb += 32) {
    v16h av = load_dualrow(arow, kb, hioff);
    v16h bv = load_dualrow(xr, kb, hioff);
    acc = __builtin_amdgcn_wmma_f32_16x16x32_f16(false, av, false, bv, (short)0, acc, false, false);
  }
  if (nvalid) {
    float* ho = headout + (size_t)(posBase + nIdx) * 16;
#pragma unroll
    for (int v = 0; v < 8; ++v) {
      int M = hi * 8 + v;
      float bias = (M < 3) ? clsb[M] : ((M < 15) ? regb[M - 3] : 0.f);
      ho[M] = acc[v] + bias;
    }
  }
}

// ---------------- sigmoid + decode ----------------

__global__ void decode_k(const float* __restrict__ headout, float* __restrict__ scores,
                         float* __restrict__ boxes) {
  int id = blockIdx.x * blockDim.x + threadIdx.x;
  if (id >= 2 * N_ANCH) return;
  int b = id / N_ANCH, g = id - b * N_ANCH;
  AnchorInfo ai = anchor_of(g);
  const float* h16 = headout + (size_t)hpos_idx(ai.l, b, ai.p) * 16;
  float cls = h16[ai.a];
  scores[id] = 1.f / (1.f + expf(-cls));
  float r0 = h16[3 + ai.a * 4 + 0], r1 = h16[3 + ai.a * 4 + 1];
  float r2 = h16[3 + ai.a * 4 + 2], r3 = h16[3 + ai.a * 4 + 3];
  float aw = ai.x2 - ai.x1, ah = ai.y2 - ai.y1;
  float acx = ai.x1 + 0.5f * aw, acy = ai.y1 + 0.5f * ah;
  float cx = r0 * aw + acx, cy = r1 * ah + acy;
  float w = expf(r2) * aw, hh = expf(r3) * ah;
  float* ob = boxes + (size_t)id * 4;
  ob[0] = cx - 0.5f * w; ob[1] = cy - 0.5f * hh;
  ob[2] = cx + 0.5f * w; ob[3] = cy + 0.5f * hh;
}

// ---------------- bitonic sorts ----------------

__device__ __forceinline__ void bitonic_u64(unsigned long long* kv, int t, int n, int nthr) {
  for (int k = 2; k <= n; k <<= 1)
    for (int j = k >> 1; j > 0; j >>= 1) {
      for (int i = t; i < n; i += nthr) {
        int ixj = i ^ j;
        if (ixj > i) {
          unsigned long long a = kv[i], c = kv[ixj];
          bool up = ((i & k) == 0);
          if (up ? (a > c) : (a < c)) { kv[i] = c; kv[ixj] = a; }
        }
      }
      __syncthreads();
    }
}

__device__ __forceinline__ void bitonic_u32(unsigned* kv, int t, int n, int nthr) {
  for (int k = 2; k <= n; k <<= 1)
    for (int j = k >> 1; j > 0; j >>= 1) {
      for (int i = t; i < n; i += nthr) {
        int ixj = i ^ j;
        if (ixj > i) {
          unsigned a = kv[i], c = kv[ixj];
          bool up = ((i & k) == 0);
          if (up ? (a > c) : (a < c)) { kv[i] = c; kv[ixj] = a; }
        }
      }
      __syncthreads();
    }
}

// ---------------- per-image sort + greedy NMS (1 workgroup per image, 128KB LDS sort) -----

__global__ __launch_bounds__(1024) void nms_k(const float* __restrict__ scores,
                                              const float* __restrict__ boxes,
                                              float* __restrict__ bsort,
                                              float* __restrict__ out) {
  int b = blockIdx.x, t = threadIdx.x;
  __shared__ unsigned long long kv[16384];
  __shared__ unsigned suppw[(N_ANCH + 31) / 32];
  __shared__ float piv[4];
  __shared__ int pflag;

  for (int i = t; i < TOPN * 4; i += 1024) out[(size_t)b * TOPN * 4 + i] = -1.f;

  for (int i = t; i < 16384; i += 1024) {
    if (i < N_ANCH) {
      unsigned fb = __float_as_uint(scores[(size_t)b * N_ANCH + i]);
      kv[i] = ((unsigned long long)(~fb) << 32) | (unsigned)i;   // descending score, stable
    } else kv[i] = ~0ull;
  }
  __syncthreads();
  bitonic_u64(kv, t, 16384, 1024);

  for (int i = t; i < N_ANCH; i += 1024) {
    int idx = (int)(kv[i] & 0xffffffffu);
    const float* bb = boxes + ((size_t)b * N_ANCH + idx) * 4;
    float* o = bsort + ((size_t)b * N_ANCH + i) * 4;
    o[0] = bb[0]; o[1] = bb[1]; o[2] = bb[2]; o[3] = bb[3];
  }
  for (int i = t; i < (N_ANCH + 31) / 32; i += 1024) suppw[i] = 0u;
  __threadfence();
  __syncthreads();

  const float* B0 = bsort + (size_t)b * N_ANCH * 4;
  for (int i = 0; i < N_ANCH - 1; ++i) {
    if (t == 0) {
      pflag = (suppw[i >> 5] >> (i & 31)) & 1;
      if (!pflag) {
        piv[0] = B0[i * 4 + 0]; piv[1] = B0[i * 4 + 1];
        piv[2] = B0[i * 4 + 2]; piv[3] = B0[i * 4 + 3];
      }
    }
    __syncthreads();
    if (!pflag) {
      float px1 = piv[0], py1 = piv[1], px2 = piv[2], py2 = piv[3];
      for (int j = i + 1 + t; j < N_ANCH; j += 1024) {
        if ((suppw[j >> 5] >> (j & 31)) & 1) continue;
        const float* bj = B0 + (size_t)j * 4;
        float iou = iou_xyxy(px1, py1, px2, py2, bj[0], bj[1], bj[2], bj[3]);
        if (iou > 0.7f) atomicOr(&suppw[j >> 5], 1u << (j & 31));
      }
    }
    __syncthreads();
  }

  if (t == 0) {
    int rank = 0;
    for (int i = 0; i < N_ANCH; ++i) {
      if (!((suppw[i >> 5] >> (i & 31)) & 1)) {
        if (rank < TOPN) {
          float* o = out + ((size_t)b * TOPN + rank) * 4;
          o[0] = B0[i * 4 + 0]; o[1] = B0[i * 4 + 1];
          o[2] = B0[i * 4 + 2]; o[3] = B0[i * 4 + 3];
        }
        rank++;
      }
    }
  }
}

// ---------------- anchor<->gt matching ----------------

__global__ void match_k(const float* __restrict__ gt, float* __restrict__ ioumax,
                        int* __restrict__ matched, unsigned* __restrict__ gmax) {
  int id = blockIdx.x * blockDim.x + threadIdx.x;
  if (id >= 2 * N_ANCH) return;
  int b = id / N_ANCH, g = id - b * N_ANCH;
  AnchorInfo ai = anchor_of(g);
  float best = -1.f; int bi = 0;
#pragma unroll 4
  for (int j = 0; j < 20; ++j) {
    const float* gj = gt + (size_t)(b * 20 + j) * 4;
    float iou = iou_xyxy(ai.x1, ai.y1, ai.x2, ai.y2, gj[0], gj[1], gj[2], gj[3]);
    if (iou > best) { best = iou; bi = j; }
    atomicMax(&gmax[b * 20 + j], __float_as_uint(iou));   // iou >= 0: bits order == float order
  }
  ioumax[id] = best;
  matched[id] = bi;
}

__global__ void markey_k(const float* __restrict__ gt, const int* __restrict__ labels,
                         const float* __restrict__ ioumax, const unsigned* __restrict__ gmax,
                         unsigned* __restrict__ keys) {
  int id = blockIdx.x * blockDim.x + threadIdx.x;
  if (id >= 2 * N_ANCH) return;
  int b = id / N_ANCH, g = id - b * N_ANCH;
  AnchorInfo ai = anchor_of(g);
  float im = ioumax[id];
  bool fg_any = false;
#pragma unroll 4
  for (int j = 0; j < 20; ++j) {
    const float* gj = gt + (size_t)(b * 20 + j) * 4;
    float thr = (labels[b * 20 + j] == -1) ? 1.f : __uint_as_float(gmax[b * 20 + j]);
    float iou = iou_xyxy(ai.x1, ai.y1, ai.x2, ai.y2, gj[0], gj[1], gj[2], gj[3]);
    if (iou == thr) fg_any = true;
  }
  int mi = 0;
  if (im > 0.7f) mi = 1;
  if (fg_any)    mi = 1;
  if (im < 0.3f) mi = -1;
  keys[id] = (((unsigned)(1 - mi)) << 14) | (unsigned)g;  // stable descending-m argsort key
}

// ---------------- sampling + BCE + smooth-L1 (1 workgroup per image) ----------------

__global__ __launch_bounds__(1024) void sample_loss_k(const unsigned* __restrict__ keys,
                                                      const float* __restrict__ headout,
                                                      const float* __restrict__ gt,
                                                      const int* __restrict__ matched,
                                                      float* __restrict__ accum) {
  int b = blockIdx.x, t = threadIdx.x;
  __shared__ unsigned sk[16384];
  for (int i = t; i < 16384; i += 1024)
    sk[i] = (i < N_ANCH) ? keys[(size_t)b * N_ANCH + i] : 0xFFFFFFFFu;
  __syncthreads();
  bitonic_u32(sk, t, 16384, 1024);

  float cN = 0.f, cD = 0.f, rN = 0.f, fgc = 0.f;
  if (t < NPOS + NNEG) {
    int pos = (t < NPOS) ? t : (N_ANCH - NNEG + (t - NPOS));
    unsigned key = sk[pos];
    int g  = (int)(key & 0x3FFFu);
    int mi = 1 - (int)(key >> 14);
    AnchorInfo ai = anchor_of(g);
    const float* h16 = headout + (size_t)hpos_idx(ai.l, b, ai.p) * 16;
    float cs = h16[ai.a];
    float z = (mi == 1) ? 1.f : 0.f;
    float valid = (mi != 0) ? 1.f : 0.f;
    float bce = fmaxf(cs, 0.f) - cs * z + log1pf(expf(-fabsf(cs)));
    cN = bce * valid; cD = valid;
    if (mi == 1) {
      int mg = matched[(size_t)b * N_ANCH + g];
      const float* gj = gt + (size_t)(b * 20 + mg) * 4;
      float aw = ai.x2 - ai.x1, ah = ai.y2 - ai.y1;
      float acx = ai.x1 + 0.5f * aw, acy = ai.y1 + 0.5f * ah;
      float gw = gj[2] - gj[0], gh = gj[3] - gj[1];
      float gcx = gj[0] + 0.5f * gw, gcy = gj[1] + 0.5f * gh;
      float tgt[4] = {(gcx - acx) / aw, (gcy - acy) / ah, logf(gw / aw), logf(gh / ah)};
#pragma unroll
      for (int c = 0; c < 4; ++c) {
        float rs = h16[3 + ai.a * 4 + c];
        float d = fabsf(rs - tgt[c]);
        rN += (d < 1.f) ? 0.5f * d * d : d - 0.5f;
      }
      fgc = 1.f;
    }
  }
  __syncthreads();
  float* rf = (float*)sk;
  rf[t] = cN; rf[1024 + t] = cD; rf[2048 + t] = rN; rf[3072 + t] = fgc;
  __syncthreads();
  for (int s = 512; s > 0; s >>= 1) {
    if (t < s) {
      rf[t] += rf[t + s];
      rf[1024 + t] += rf[1024 + t + s];
      rf[2048 + t] += rf[2048 + t + s];
      rf[3072 + t] += rf[3072 + t + s];
    }
    __syncthreads();
  }
  if (t == 0) {
    atomicAdd(&accum[0], rf[0]);
    atomicAdd(&accum[1], rf[1024]);
    atomicAdd(&accum[2], rf[2048]);
    atomicAdd(&accum[3], rf[3072]);
  }
}

__global__ void final_k(const float* __restrict__ accum, float* __restrict__ out) {
  out[16000] = accum[0] / fmaxf(accum[1], 1.f);
  out[16001] = accum[2] / fmaxf(accum[3] * 4.f, 1.f);
}

// ---------------- launcher ----------------

extern "C" void kernel_launch(void* const* d_in, const int* in_sizes, int n_in,
                              void* d_out, int out_size, void* d_ws, size_t ws_size,
                              hipStream_t stream) {
  const float* feats[5] = {(const float*)d_in[0], (const float*)d_in[1], (const float*)d_in[2],
                           (const float*)d_in[3], (const float*)d_in[4]};
  const float* conv_w = (const float*)d_in[5];
  const float* conv_b = (const float*)d_in[6];
  const float* cls_w  = (const float*)d_in[7];
  const float* cls_b  = (const float*)d_in[8];
  const float* reg_w  = (const float*)d_in[9];
  const float* reg_b  = (const float*)d_in[10];
  const float* gt     = (const float*)d_in[11];
  const int*   labels = (const int*)d_in[12];
  float* out = (float*)d_out;

  char* ws = (char*)d_ws;
  size_t off = 0;
  auto take = [&](size_t bytes) { char* p = ws + off; off = (off + bytes + 255) & ~size_t(255); return p; };
  _Float16* wA      = (_Float16*)take(256 * 2304 * 2);
  _Float16* wCR     = (_Float16*)take(16 * 256 * 2);
  _Float16* xcol    = (_Float16*)take((size_t)6138 * 2304 * 2);   // im2col, 28 MB
  _Float16* xbuf    = (_Float16*)take((size_t)6138 * 256 * 2);
  float*    headout = (float*)take((size_t)6138 * 16 * 4);
  float*    scores  = (float*)take((size_t)2 * N_ANCH * 4);
  float*    boxes   = (float*)take((size_t)2 * N_ANCH * 16);
  float*    bsort   = (float*)take((size_t)2 * N_ANCH * 16);
  float*    ioumax  = (float*)take((size_t)2 * N_ANCH * 4);
  int*      matched = (int*)take((size_t)2 * N_ANCH * 4);
  unsigned* gmax    = (unsigned*)take(40 * 4);
  unsigned* keys    = (unsigned*)take((size_t)2 * N_ANCH * 4);
  float*    accum   = (float*)take(4 * 4);
  (void)ws_size; (void)in_sizes; (void)n_in; (void)out_size;

  init_k<<<1, 64, 0, stream>>>(gmax, accum);
  cvtw_k<<<(256 * 2304 + 255) / 256, 256, 0, stream>>>(conv_w, cls_w, reg_w, wA, wCR);

  const int Hs[5] = {48, 24, 12, 6, 3};
  const int PB[5] = {0, 4608, 5760, 6048, 6120};
  for (int l = 0; l < 5; ++l) {
    int H = Hs[l], SPL = H * H;
    long long total = (long long)2 * SPL * 2304;
    int blocks = (int)((total + 255) / 256);
    if (blocks > 4096) blocks = 4096;
    im2col_k<<<blocks, 256, 0, stream>>>(feats[l], xcol, H, H, PB[l]);
  }
  for (int l = 0; l < 5; ++l) {
    int H = Hs[l], SPL = H * H, N = 2 * SPL;
    int tiles = 4 * ((N + 15) / 16);            // 4 M-groups (64 rows) x N-tiles, one wave each
    int blocks = (tiles + 7) / 8;               // 256 threads = 8 wave32
    conv3_wmma<<<blocks, 256, 0, stream>>>(xcol, wA, conv_b, xbuf, SPL, PB[l]);
  }
  for (int l = 0; l < 5; ++l) {
    int H = Hs[l], SPL = H * H, N = 2 * SPL;
    int tiles = (N + 15) / 16;
    int blocks = (tiles + 7) / 8;
    head_wmma<<<blocks, 256, 0, stream>>>(xbuf, wCR, cls_b, reg_b, headout, SPL, PB[l]);
  }
  decode_k<<<(2 * N_ANCH + 255) / 256, 256, 0, stream>>>(headout, scores, boxes);
  nms_k<<<2, 1024, 0, stream>>>(scores, boxes, bsort, out);
  match_k<<<(2 * N_ANCH + 255) / 256, 256, 0, stream>>>(gt, ioumax, matched, gmax);
  markey_k<<<(2 * N_ANCH + 255) / 256, 256, 0, stream>>>(gt, labels, ioumax, gmax, keys);
  sample_loss_k<<<2, 1024, 0, stream>>>(keys, headout, gt, matched, accum);
  final_k<<<1, 1, 0, stream>>>(accum, out);
}
